// RelSelfAttention2DShaw_26886495273432
// MI455X (gfx1250) — matrix operand order repaired and checked by reading, not compiled, not run on hardware
//
#include <hip/hip_runtime.h>
#include <hip/hip_bf16.h>

typedef __bf16 bf16_t;
typedef __attribute__((ext_vector_type(16))) __bf16 v16bf;
typedef __attribute__((ext_vector_type(8)))  __bf16 v8bf;
typedef __attribute__((ext_vector_type(8)))  float  v8f;
typedef __attribute__((ext_vector_type(4)))  float  v4f;

#define DEVI __device__ __forceinline__

// Problem constants
constexpr int Bq = 1024, Tt = 64, Dm = 1024, Hh = 16, DH = 64, NREL = 225;

// ---------------- workspace layout (bytes) ----------------
constexpr size_t SZ_W   = (size_t)Dm * Dm * 2;                 // 2 MB each
constexpr size_t SZ_QKV = (size_t)Bq * Hh * Tt * DH * 2;       // 128 MB each
constexpr size_t SZ_REL = (size_t)Hh * Tt * Tt * DH * 2;       // 8 MB each
constexpr size_t SZ_QRK = (size_t)Hh * Tt * Tt * 4;            // 256 KB
constexpr size_t SZ_L   = (size_t)Bq * Hh * Tt * Tt * 4;       // 256 MB
constexpr size_t SZ_AT  = (size_t)Bq * Hh * Tt * Tt * 2;       // 128 MB
constexpr size_t SZ_O   = (size_t)Bq * Tt * Dm * 2;            // 128 MB

constexpr size_t OFF_WQ  = 0;
constexpr size_t OFF_WK  = OFF_WQ + SZ_W;
constexpr size_t OFF_WV  = OFF_WK + SZ_W;
constexpr size_t OFF_WO  = OFF_WV + SZ_W;
constexpr size_t OFF_Q   = OFF_WO + SZ_W;
constexpr size_t OFF_K   = OFF_Q  + SZ_QKV;
constexpr size_t OFF_V   = OFF_K  + SZ_QKV;
constexpr size_t OFF_RQ  = OFF_V  + SZ_QKV;
constexpr size_t OFF_RK  = OFF_RQ + SZ_REL;
constexpr size_t OFF_RV  = OFF_RK + SZ_REL;
constexpr size_t OFF_QRK = OFF_RV + SZ_REL;
constexpr size_t OFF_L   = OFF_QRK + SZ_QRK;
constexpr size_t OFF_AT  = OFF_L  + SZ_L;
constexpr size_t OFF_O   = OFF_AT + SZ_AT;

// ---------------- WMMA fragment helpers (wave32, contiguous k order used here) -------
// A matrix 16x32 bf16: lane(0..15)=row M, holds K = hi*16 + {0..15} contiguous
DEVI v16bf ld_fragA(const bf16_t* p /* row*stride + kb + hi*16 */) {
  v8bf lo = *(const v8bf*)(p);
  v8bf hh = *(const v8bf*)(p + 8);
  v16bf r;
#pragma unroll
  for (int i = 0; i < 8; ++i) { r[i] = lo[i]; r[i + 8] = hh[i]; }
  return r;
}
// B matrix 32x16 bf16 (stored as 16 n-rows x K cols):
// lane(0..15)=col N, holds K = hi*16 + {0..15} contiguous
DEVI v16bf ld_fragB(const bf16_t* p /* nrow*stride + kb + hi*16 */) {
  v8bf lo = *(const v8bf*)(p);
  v8bf hh = *(const v8bf*)(p + 8);
  v16bf r;
#pragma unroll
  for (int i = 0; i < 8; ++i) { r[i] = lo[i]; r[i + 8] = hh[i]; }
  return r;
}
DEVI v8f wmma_bf16(v16bf a, v16bf b, v8f c) {
  return __builtin_amdgcn_wmma_f32_16x16x32_bf16(false, a, false, b, (short)0, c,
                                                 false, false);
}

// ---------------- K0: f32 -> bf16 convert ----------------
__global__ void k_cvt_bf16(const float* __restrict__ src, bf16_t* __restrict__ dst,
                           int n) {
  int i = blockIdx.x * blockDim.x + threadIdx.x;
  if (i < n) dst[i] = (bf16_t)src[i];
}

// ---------------- K1: gather rel tensors + relq_relk ----------------
__global__ void k_gather_rel(const float* __restrict__ rq, const float* __restrict__ rk,
                             const float* __restrict__ rv, const int* __restrict__ ridx,
                             bf16_t* __restrict__ rqg, bf16_t* __restrict__ rkg,
                             bf16_t* __restrict__ rvg, float* __restrict__ relqk) {
  int i = blockIdx.x * blockDim.x + threadIdx.x;  // (h,t,s), 65536 total
  if (i >= Hh * Tt * Tt) return;
  int s = i & 63, t = (i >> 6) & 63, h = i >> 12;
  int idx = ridx[t * 64 + s];
  const float* pq = rq + ((size_t)h * NREL + idx) * DH;
  const float* pk = rk + ((size_t)h * NREL + idx) * DH;
  const float* pv = rv + ((size_t)h * NREL + idx) * DH;
  bf16_t* oq = rqg + (size_t)i * DH;
  bf16_t* ok = rkg + (size_t)i * DH;
  bf16_t* ov = rvg + (size_t)i * DH;
  float acc = 0.f;
#pragma unroll 8
  for (int d = 0; d < DH; ++d) {
    float a = pq[d], b = pk[d];
    acc += a * b;
    oq[d] = (bf16_t)a;
    ok[d] = (bf16_t)b;
    ov[d] = (bf16_t)pv[d];
  }
  relqk[i] = acc;
}

// ---------------- K2: QKV projection GEMM (M=65536, N=1024, K=1024) ------
// 128x64 C tile per WG; 8 waves, each wave = 16 rows x 64 cols (4 accs)
// out layout: [(b*H + h)*T + t]*DH + d  (contiguous 64x64 per (b,h))
__global__ __launch_bounds__(256) void k_gemm_qkv(
    const float* __restrict__ X, const bf16_t* __restrict__ Wb,
    const float* __restrict__ bias, bf16_t* __restrict__ out) {
  __shared__ __align__(16) bf16_t As[128 * 32];  // 8 KB
  __shared__ __align__(16) bf16_t Bs[64 * 32];   // 4 KB
  const int tid = threadIdx.x, lane = tid & 31, wave = tid >> 5;
  const int lo = lane & 15, hi = lane >> 4, wm = wave;  // wm 0..7
  const int mt = blockIdx.x, nt = blockIdx.y;
  const int ar = tid >> 1, ac = (tid & 1) * 16;  // A: 128 rows x 32, 16 els/thread
  const int br = tid >> 2, bc = (tid & 3) * 8;   // B: 64 rows x 32,  8 els/thread
  const float*  xp = X  + (size_t)(mt * 128 + ar) * Dm + ac;
  const bf16_t* wp = Wb + (size_t)(nt * 64 + br) * Dm + bc;
  v8f acc[4] = {};
  for (int kb = 0; kb < Dm; kb += 32) {
    v4f x0 = *(const v4f*)(xp + kb);
    v4f x1 = *(const v4f*)(xp + kb + 4);
    v4f x2 = *(const v4f*)(xp + kb + 8);
    v4f x3 = *(const v4f*)(xp + kb + 12);
    if (kb + 128 < Dm) {
      __builtin_prefetch(xp + kb + 128, 0, 0);
      __builtin_prefetch(wp + kb + 128, 0, 0);
    }
    v8bf a0, a1;
#pragma unroll
    for (int i = 0; i < 4; ++i) {
      a0[i] = (bf16_t)x0[i]; a0[i + 4] = (bf16_t)x1[i];
      a1[i] = (bf16_t)x2[i]; a1[i + 4] = (bf16_t)x3[i];
    }
    *(v8bf*)&As[ar * 32 + ac]     = a0;
    *(v8bf*)&As[ar * 32 + ac + 8] = a1;
    *(v8bf*)&Bs[br * 32 + bc]     = *(const v8bf*)(wp + kb);
    __syncthreads();
    v16bf a = ld_fragA(&As[(wm * 16 + lo) * 32 + hi * 16]);
#pragma unroll
    for (int c = 0; c < 4; ++c) {
      v16bf b = ld_fragB(&Bs[(c * 16 + lo) * 32 + hi * 16]);
      acc[c] = wmma_bf16(a, b, acc[c]);
    }
    __syncthreads();
  }
#pragma unroll
  for (int c = 0; c < 4; ++c) {
#pragma unroll
    for (int j = 0; j < 8; ++j) {
      int m = mt * 128 + wm * 16 + j + 8 * hi;
      int b = m >> 6, t = m & 63;
      int n = nt * 64 + c * 16 + lo;
      out[(((size_t)b * Hh + (n >> 6)) * Tt + t) * DH + (n & 63)] =
          (bf16_t)(acc[c][j] + bias[n]);
    }
  }
}

// ---------------- K3: L = q_relk + relq_relk  (per (h,t): M=B,N=T,K=DH) --
// 128x64 C tile per WG (M = batch b)
__global__ __launch_bounds__(256) void k_qrelk(
    const bf16_t* __restrict__ Q, const bf16_t* __restrict__ RK,
    const float* __restrict__ relqk, float* __restrict__ L) {
  __shared__ __align__(16) bf16_t As[128 * 64];  // 16 KB
  __shared__ __align__(16) bf16_t Bs[64 * 64];   //  8 KB
  const int tid = threadIdx.x, lane = tid & 31, wave = tid >> 5;
  const int lo = lane & 15, hi = lane >> 4, wm = wave;
  const int btile = blockIdx.x, ht = blockIdx.y;  // ht = h*64 + t
  const int h = ht >> 6, t = ht & 63;
  {
    int ar = tid >> 1, ac = (tid & 1) * 32;  // 128 rows x 64, 32 els/thread
    size_t qa = (((size_t)(btile * 128 + ar) * Hh + h) * Tt + t) * DH + ac;
#pragma unroll
    for (int i = 0; i < 4; ++i)
      *(v8bf*)&As[ar * 64 + ac + i * 8] = *(const v8bf*)(Q + qa + i * 8);
    int brr = tid >> 2, bcc = (tid & 3) * 16;  // 64 x 64, 16 els/thread
    size_t ba = ((size_t)ht * Tt + brr) * DH + bcc;
    *(v8bf*)&Bs[brr * 64 + bcc]     = *(const v8bf*)(RK + ba);
    *(v8bf*)&Bs[brr * 64 + bcc + 8] = *(const v8bf*)(RK + ba + 8);
  }
  __syncthreads();
  v8f acc[4] = {};
#pragma unroll
  for (int kb = 0; kb < 64; kb += 32) {
    v16bf a = ld_fragA(&As[(wm * 16 + lo) * 64 + kb + hi * 16]);
#pragma unroll
    for (int c = 0; c < 4; ++c) {
      v16bf b = ld_fragB(&Bs[(c * 16 + lo) * 64 + kb + hi * 16]);
      acc[c] = wmma_bf16(a, b, acc[c]);
    }
  }
#pragma unroll
  for (int c = 0; c < 4; ++c) {
#pragma unroll
    for (int j = 0; j < 8; ++j) {
      int b = btile * 128 + wm * 16 + j + 8 * hi;
      int s = c * 16 + lo;
      size_t base = (((size_t)b * Hh + h) * Tt + t) * Tt;
      L[base + s] = acc[c][j] + relqk[(size_t)ht * Tt + s];
    }
  }
}

// ---------------- K4: L += relq_k  (per (h,s): M=B, N=T(t), K=DH) --------
__global__ __launch_bounds__(256) void k_relqk_add(
    const bf16_t* __restrict__ Kb, const bf16_t* __restrict__ RQ,
    float* __restrict__ L) {
  __shared__ __align__(16) bf16_t As[128 * 64];
  __shared__ __align__(16) bf16_t Bs[64 * 64];
  const int tid = threadIdx.x, lane = tid & 31, wave = tid >> 5;
  const int lo = lane & 15, hi = lane >> 4, wm = wave;
  const int btile = blockIdx.x, hs = blockIdx.y;  // hs = h*64 + s
  const int h = hs >> 6, s = hs & 63;
  {
    int ar = tid >> 1, ac = (tid & 1) * 32;
    size_t ka = (((size_t)(btile * 128 + ar) * Hh + h) * Tt + s) * DH + ac;
#pragma unroll
    for (int i = 0; i < 4; ++i)
      *(v8bf*)&As[ar * 64 + ac + i * 8] = *(const v8bf*)(Kb + ka + i * 8);
    // B rows = t: rq[h][t][s][:] (row stride T*DH)
    int brr = tid >> 2, bcc = (tid & 3) * 16;
    size_t ba = (((size_t)h * Tt + brr) * Tt + s) * DH + bcc;
    *(v8bf*)&Bs[brr * 64 + bcc]     = *(const v8bf*)(RQ + ba);
    *(v8bf*)&Bs[brr * 64 + bcc + 8] = *(const v8bf*)(RQ + ba + 8);
  }
  __syncthreads();
  v8f acc[4] = {};
#pragma unroll
  for (int kb = 0; kb < 64; kb += 32) {
    v16bf a = ld_fragA(&As[(wm * 16 + lo) * 64 + kb + hi * 16]);
#pragma unroll
    for (int c = 0; c < 4; ++c) {
      v16bf b = ld_fragB(&Bs[(c * 16 + lo) * 64 + kb + hi * 16]);
      acc[c] = wmma_bf16(a, b, acc[c]);
    }
  }
#pragma unroll
  for (int c = 0; c < 4; ++c) {
#pragma unroll
    for (int j = 0; j < 8; ++j) {
      int b = btile * 128 + wm * 16 + j + 8 * hi;
      int t = c * 16 + lo;
      L[(((size_t)b * Hh + h) * Tt + t) * Tt + s] += acc[c][j];
    }
  }
}

// ---------------- K5: fused attention per (b,h) --------------------------
__global__ __launch_bounds__(256) void k_attn(
    const bf16_t* __restrict__ Q, const bf16_t* __restrict__ Kb,
    const bf16_t* __restrict__ V, const float* __restrict__ L,
    bf16_t* __restrict__ attnG, bf16_t* __restrict__ O) {
  __shared__ __align__(16) bf16_t Qs[64 * 64];
  __shared__ __align__(16) bf16_t Ks[64 * 64];
  __shared__ __align__(16) bf16_t Vs[64 * 64];
  __shared__ __align__(16) float  Sf[64 * 64];
  __shared__ __align__(16) bf16_t Ps[64 * 64];
  const int bh = blockIdx.x;  // b*16 + h
  const int tid = threadIdx.x, lane = tid & 31, wave = tid >> 5;
  const int lo = lane & 15, hi = lane >> 4, wm = wave >> 1, wn = wave & 1;
  const size_t base = (size_t)bh * (Tt * DH);  // contiguous 64x64 block
  {
    int off = tid * 16;
    *(v8bf*)&Qs[off]     = *(const v8bf*)(Q + base + off);
    *(v8bf*)&Qs[off + 8] = *(const v8bf*)(Q + base + off + 8);
    *(v8bf*)&Ks[off]     = *(const v8bf*)(Kb + base + off);
    *(v8bf*)&Ks[off + 8] = *(const v8bf*)(Kb + base + off + 8);
    *(v8bf*)&Vs[off]     = *(const v8bf*)(V + base + off);
    *(v8bf*)&Vs[off + 8] = *(const v8bf*)(V + base + off + 8);
  }
  __syncthreads();
  // S = Q @ K^T
  v8f acc0 = {}, acc1 = {};
#pragma unroll
  for (int kb = 0; kb < 64; kb += 32) {
    v16bf a  = ld_fragA(&Qs[(wm * 16 + lo) * 64 + kb + hi * 16]);
    v16bf b0 = ld_fragB(&Ks[(wn * 32 + lo) * 64 + kb + hi * 16]);
    v16bf b1 = ld_fragB(&Ks[(wn * 32 + 16 + lo) * 64 + kb + hi * 16]);
    acc0 = wmma_bf16(a, b0, acc0);
    acc1 = wmma_bf16(a, b1, acc1);
  }
  const float scale = 0.125f;  // 1/sqrt(64)
#pragma unroll
  for (int j = 0; j < 8; ++j) {
    int t = wm * 16 + j + 8 * hi;
    int s0 = wn * 32 + lo, s1 = s0 + 16;
    Sf[t * 64 + s0] = (acc0[j] + L[base + t * 64 + s0]) * scale;
    Sf[t * 64 + s1] = (acc1[j] + L[base + t * 64 + s1]) * scale;
  }
  __syncthreads();
  // softmax over s, one thread per row t
  if (tid < 64) {
    int t = tid;
    float mx = -1e30f;
    for (int s = 0; s < 64; ++s) mx = fmaxf(mx, Sf[t * 64 + s]);
    float sum = 0.f;
    for (int s = 0; s < 64; ++s) {
      float e = __expf(Sf[t * 64 + s] - mx);
      Sf[t * 64 + s] = e;
      sum += e;
    }
    float inv = 1.f / sum;
    for (int s = 0; s < 64; ++s) {
      bf16_t p = (bf16_t)(Sf[t * 64 + s] * inv);
      Ps[t * 64 + s] = p;
      attnG[base + t * 64 + s] = p;
    }
  }
  __syncthreads();
  // O1 = P @ V^T
  v8f o0 = {}, o1 = {};
#pragma unroll
  for (int kb = 0; kb < 64; kb += 32) {
    v16bf a  = ld_fragA(&Ps[(wm * 16 + lo) * 64 + kb + hi * 16]);
    v16bf b0 = ld_fragB(&Vs[(wn * 32 + lo) * 64 + kb + hi * 16]);
    v16bf b1 = ld_fragB(&Vs[(wn * 32 + 16 + lo) * 64 + kb + hi * 16]);
    o0 = wmma_bf16(a, b0, o0);
    o1 = wmma_bf16(a, b1, o1);
  }
  const int b = bh >> 4, h = bh & 15;
#pragma unroll
  for (int j = 0; j < 8; ++j) {
    int t = wm * 16 + j + 8 * hi;
    int d0 = wn * 32 + lo, d1 = d0 + 16;
    size_t oi = ((size_t)b * Tt + t) * Dm + h * DH;
    O[oi + d0] = (bf16_t)o0[j];
    O[oi + d1] = (bf16_t)o1[j];
  }
}

// ---------------- K6: O += attn @ rv  (per (h,t): M=B, N=DH, K=T) --------
__global__ __launch_bounds__(256) void k_attn_rv(
    const bf16_t* __restrict__ attnG, const bf16_t* __restrict__ RV,
    bf16_t* __restrict__ O) {
  __shared__ __align__(16) bf16_t As[128 * 64];
  __shared__ __align__(16) bf16_t Bs[64 * 64];
  const int tid = threadIdx.x, lane = tid & 31, wave = tid >> 5;
  const int lo = lane & 15, hi = lane >> 4, wm = wave;
  const int btile = blockIdx.x, ht = blockIdx.y;  // ht = h*64 + t
  const int h = ht >> 6, t = ht & 63;
  {
    int ar = tid >> 1, ac = (tid & 1) * 32;
    size_t aa = (((size_t)(btile * 128 + ar) * Hh + h) * Tt + t) * Tt + ac;
#pragma unroll
    for (int i = 0; i < 4; ++i)
      *(v8bf*)&As[ar * 64 + ac + i * 8] = *(const v8bf*)(attnG + aa + i * 8);
    // B operand needs n-rows = d: stage rv transposed -> Bs[d][s]
    int s = tid >> 2, dbase = (tid & 3) * 16;
    size_t ba = ((size_t)ht * Tt + s) * DH + dbase;  // rv[s][dbase..]
    v8bf r0 = *(const v8bf*)(RV + ba);
    v8bf r1 = *(const v8bf*)(RV + ba + 8);
#pragma unroll
    for (int i = 0; i < 8; ++i) {
      Bs[(dbase + i) * 64 + s]     = r0[i];
      Bs[(dbase + 8 + i) * 64 + s] = r1[i];
    }
  }
  __syncthreads();
  v8f acc[4] = {};
#pragma unroll
  for (int kb = 0; kb < 64; kb += 32) {  // k = s
    v16bf a = ld_fragA(&As[(wm * 16 + lo) * 64 + kb + hi * 16]);
#pragma unroll
    for (int c = 0; c < 4; ++c) {
      v16bf b = ld_fragB(&Bs[(c * 16 + lo) * 64 + kb + hi * 16]);
      acc[c] = wmma_bf16(a, b, acc[c]);
    }
  }
#pragma unroll
  for (int c = 0; c < 4; ++c) {
#pragma unroll
    for (int j = 0; j < 8; ++j) {
      int b = btile * 128 + wm * 16 + j + 8 * hi;
      int d = c * 16 + lo;
      size_t oi = ((size_t)b * Tt + t) * Dm + h * DH + d;
      O[oi] = (bf16_t)((float)O[oi] + acc[c][j]);
    }
  }
}

// ---------------- K7: output projection GEMM -> f32 d_out ----------------
__global__ __launch_bounds__(256) void k_gemm_out(
    const bf16_t* __restrict__ A, const bf16_t* __restrict__ Wb,
    const float* __restrict__ bias, float* __restrict__ out) {
  __shared__ __align__(16) bf16_t As[128 * 32];
  __shared__ __align__(16) bf16_t Bs[64 * 32];
  const int tid = threadIdx.x, lane = tid & 31, wave = tid >> 5;
  const int lo = lane & 15, hi = lane >> 4, wm = wave;
  const int mt = blockIdx.x, nt = blockIdx.y;
  const int ar = tid >> 1, ac = (tid & 1) * 16;
  const int br = tid >> 2, bc = (tid & 3) * 8;
  const bf16_t* apg = A  + (size_t)(mt * 128 + ar) * Dm + ac;
  const bf16_t* wp  = Wb + (size_t)(nt * 64 + br) * Dm + bc;
  v8f acc[4] = {};
  for (int kb = 0; kb < Dm; kb += 32) {
    if (kb + 128 < Dm) {
      __builtin_prefetch(apg + kb + 128, 0, 0);
      __builtin_prefetch(wp + kb + 128, 0, 0);
    }
    *(v8bf*)&As[ar * 32 + ac]     = *(const v8bf*)(apg + kb);
    *(v8bf*)&As[ar * 32 + ac + 8] = *(const v8bf*)(apg + kb + 8);
    *(v8bf*)&Bs[br * 32 + bc]     = *(const v8bf*)(wp + kb);
    __syncthreads();
    v16bf a = ld_fragA(&As[(wm * 16 + lo) * 32 + hi * 16]);
#pragma unroll
    for (int c = 0; c < 4; ++c) {
      v16bf b = ld_fragB(&Bs[(c * 16 + lo) * 32 + hi * 16]);
      acc[c] = wmma_bf16(a, b, acc[c]);
    }
    __syncthreads();
  }
#pragma unroll
  for (int c = 0; c < 4; ++c) {
#pragma unroll
    for (int j = 0; j < 8; ++j) {
      int m = mt * 128 + wm * 16 + j + 8 * hi;
      int n = nt * 64 + c * 16 + lo;
      out[(size_t)m * Dm + n] = acc[c][j] + bias[n];
    }
  }
}

// ---------------- host side ----------------
extern "C" void kernel_launch(void* const* d_in, const int* in_sizes, int n_in,
                              void* d_out, int out_size, void* d_ws, size_t ws_size,
                              hipStream_t stream) {
  (void)in_sizes; (void)n_in; (void)out_size; (void)ws_size;
  const float* x    = (const float*)d_in[0];
  const float* Wq_w = (const float*)d_in[1];
  const float* Wq_b = (const float*)d_in[2];
  const float* Wk_w = (const float*)d_in[3];
  const float* Wk_b = (const float*)d_in[4];
  const float* Wv_w = (const float*)d_in[5];
  const float* Wv_b = (const float*)d_in[6];
  const float* Wo_w = (const float*)d_in[7];
  const float* Wo_b = (const float*)d_in[8];
  const float* rel_q = (const float*)d_in[9];
  const float* rel_k = (const float*)d_in[10];
  const float* rel_v = (const float*)d_in[11];
  const int*   ridx  = (const int*)d_in[12];

  char* ws = (char*)d_ws;
  bf16_t* wqb = (bf16_t*)(ws + OFF_WQ);
  bf16_t* wkb = (bf16_t*)(ws + OFF_WK);
  bf16_t* wvb = (bf16_t*)(ws + OFF_WV);
  bf16_t* wob = (bf16_t*)(ws + OFF_WO);
  bf16_t* qb  = (bf16_t*)(ws + OFF_Q);
  bf16_t* kb  = (bf16_t*)(ws + OFF_K);
  bf16_t* vb  = (bf16_t*)(ws + OFF_V);
  bf16_t* rqg = (bf16_t*)(ws + OFF_RQ);
  bf16_t* rkg = (bf16_t*)(ws + OFF_RK);
  bf16_t* rvg = (bf16_t*)(ws + OFF_RV);
  float*  qrk = (float*)(ws + OFF_QRK);
  float*  L   = (float*)(ws + OFF_L);
  bf16_t* at  = (bf16_t*)(ws + OFF_AT);
  bf16_t* O   = (bf16_t*)(ws + OFF_O);
  float*  out = (float*)d_out;

  const int nW = Dm * Dm;
  k_cvt_bf16<<<nW / 256, 256, 0, stream>>>(Wq_w, wqb, nW);
  k_cvt_bf16<<<nW / 256, 256, 0, stream>>>(Wk_w, wkb, nW);
  k_cvt_bf16<<<nW / 256, 256, 0, stream>>>(Wv_w, wvb, nW);
  k_cvt_bf16<<<nW / 256, 256, 0, stream>>>(Wo_w, wob, nW);

  k_gather_rel<<<(Hh * Tt * Tt) / 256, 256, 0, stream>>>(rel_q, rel_k, rel_v, ridx,
                                                         rqg, rkg, rvg, qrk);

  dim3 gQKV(Bq * Tt / 128, Dm / 64);
  k_gemm_qkv<<<gQKV, 256, 0, stream>>>(x, wqb, Wq_b, qb);
  k_gemm_qkv<<<gQKV, 256, 0, stream>>>(x, wkb, Wk_b, kb);
  k_gemm_qkv<<<gQKV, 256, 0, stream>>>(x, wvb, Wv_b, vb);

  dim3 gRel(Bq / 128, Hh * Tt);
  k_qrelk<<<gRel, 256, 0, stream>>>(qb, rkg, qrk, L);
  k_relqk_add<<<gRel, 256, 0, stream>>>(kb, rqg, L);

  k_attn<<<Bq * Hh, 256, 0, stream>>>(qb, kb, vb, L, at, O);

  k_attn_rv<<<gRel, 256, 0, stream>>>(at, rvg, O);

  dim3 gOut(Bq * Tt / 128, Dm / 64);
  k_gemm_out<<<gOut, 256, 0, stream>>>(O, wob, Wo_b, out);
}
